// NeuralLogicRec_30855045054522
// MI455X (gfx1250) — compile-verified, matrix operand order checked by reading, not run on hardware
//
#include <hip/hip_runtime.h>
#include <hip/hip_bf16.h>
#include <math.h>

typedef __attribute__((ext_vector_type(16))) _Float16 v16h;
typedef __attribute__((ext_vector_type(8)))  float    v8f;

#define B_USERS 256
#define N_ITEMS 10000
#define D_EMB   64
#define HID     32
#define ITEM_CHUNK 80
#define PAIR_ITERS 10   // 80 items / (4 slots * 2 items)

// ---------------------------------------------------------------------------
// Workspace layout (bytes):
//   0       : u_part  f32 [2][256][32]      (65536 B)   b0 folded in
//   65536   : norm_eu f32 [256][64]         (65536 B)
//   131072  : packedW f16 [2][2][2][32][16] (8192 B)    A-fragment layout
//   139264  : i_part  f32 [2][10000][32]    (2560000 B)
// ---------------------------------------------------------------------------

// u_part[h][b][k] = b0[k] + sum_d eu[b][d] * w0[d][k]   (w0 rows 0..63)
__global__ void k_user_prep(const int* __restrict__ users,
                            const float* __restrict__ user_emb,
                            const float* __restrict__ lw0, const float* __restrict__ lb0,
                            const float* __restrict__ rw0, const float* __restrict__ rb0,
                            float* __restrict__ u_part) {
    int tid = blockIdx.x * blockDim.x + threadIdx.x;   // 64 blocks x 256 = 16384
    int b = tid >> 6;
    int h = (tid >> 5) & 1;
    int k = tid & 31;
    if (b >= B_USERS) return;
    const float* w0 = h ? rw0 : lw0;
    const float* b0 = h ? rb0 : lb0;
    const float* eu = user_emb + (long)users[b] * D_EMB;
    float acc = b0[k];
    #pragma unroll
    for (int d = 0; d < D_EMB; ++d) acc += eu[d] * w0[d * HID + k];
    u_part[(h * B_USERS + b) * HID + k] = acc;
}

// normalized user embeddings for cosine similarity
__global__ void k_norm(const int* __restrict__ users,
                       const float* __restrict__ user_emb,
                       float* __restrict__ norm_eu) {
    int b = threadIdx.x;  // 1 block x 256
    const float* eu = user_emb + (long)users[b] * D_EMB;
    float ss = 0.f;
    #pragma unroll
    for (int d = 0; d < D_EMB; ++d) ss += eu[d] * eu[d];
    float r = rsqrtf(fmaxf(ss, 1e-12f));
    #pragma unroll
    for (int d = 0; d < D_EMB; ++d) norm_eu[b * D_EMB + d] = eu[d] * r;
}

// i_part[h][i][k] = sum_d ie[i][d] * w0[64+d][k]   (no bias; bias lives in u_part)
__global__ void k_ipart(const float* __restrict__ item_emb,
                        const float* __restrict__ lw0,
                        const float* __restrict__ rw0,
                        float* __restrict__ i_part) {
    int tid = blockIdx.x * blockDim.x + threadIdx.x;   // 2500 blocks x 256 = 640000
    int i = tid >> 6;
    int h = (tid >> 5) & 1;
    int k = tid & 31;
    if (i >= N_ITEMS) return;
    const float* w0 = (h ? rw0 : lw0) + D_EMB * HID;   // rows 64..127
    const float* x = item_emb + (size_t)i * D_EMB;
    float acc = 0.f;
    #pragma unroll
    for (int d = 0; d < D_EMB; ++d) acc += x[d] * w0[d * HID + k];
    i_part[(size_t)(h * N_ITEMS + i) * HID + k] = acc;
}

// Pre-swizzle W1/W2 of both heads into WMMA A-fragment layout:
// A chunk c holds (W^T)[16c+m][k] = W[k][16c+m].
// f16 A layout: lane l (M = l%16), element e: k = (e<8 ? e : 16+e-8) + 8*(l>>4)
__global__ void k_pack(const float* __restrict__ lw1, const float* __restrict__ lw2,
                       const float* __restrict__ rw1, const float* __restrict__ rw2,
                       _Float16* __restrict__ packed) {
    int tid = threadIdx.x;                 // 1 block x 256
    int lane  = tid & 31;
    int chunk = (tid >> 5) & 1;
    int layer = (tid >> 6) & 1;
    int head  = (tid >> 7) & 1;
    const float* w = head ? (layer ? rw2 : rw1) : (layer ? lw2 : lw1);
    int koff = ((lane >> 4) & 1) * 8;
    int mm = chunk * 16 + (lane & 15);
    #pragma unroll
    for (int e = 0; e < 16; ++e) {
        int k = (e < 8) ? (koff + e) : (16 + koff + (e - 8));
        packed[tid * 16 + e] = (_Float16)w[k * HID + mm];
    }
}

// -------- main pair-head kernel -------------------------------------------
// wave = (item-slot, head); per iteration: 2 adjacent items x 16 users,
// 4 independent WMMAs per layer to hide WMMA->VALU hazard NOPs; i_part
// loads double-buffered one iteration ahead.
__global__ void __launch_bounds__(256)
k_pair(const float* __restrict__ u_part, const float* __restrict__ i_part,
       const _Float16* __restrict__ packedW,
       const float* __restrict__ lb1, const float* __restrict__ lb2,
       const float* __restrict__ lw3, const float* __restrict__ lb3,
       const float* __restrict__ rb1, const float* __restrict__ rb2,
       const float* __restrict__ rw3, const float* __restrict__ rb3,
       float* __restrict__ likes, float* __restrict__ rated) {
    const int lane = threadIdx.x & 31;
    const int wave = threadIdx.x >> 5;
    const int head = wave & 1;
    const int slot = wave >> 1;          // 0..3
    const int n    = lane & 15;          // pair column (user within tile)
    const int koff = (lane >> 4) << 3;   // 0 or 8
    const int utile = blockIdx.x * 16;

    const float* bias1 = head ? rb1 : lb1;
    const float* bias2 = head ? rb2 : lb2;
    const float* w3    = head ? rw3 : lw3;
    const float  b3    = (head ? rb3 : lb3)[0];
    float* out = head ? rated : likes;

    // per-lane user partial (b0 included), k in {koff..koff+7} U {16+koff..}
    const float* up = u_part + (size_t)(head * B_USERS + utile + n) * HID;
    float u[16];
    #pragma unroll
    for (int j = 0; j < 8; ++j) { u[j] = up[koff + j]; u[8 + j] = up[16 + koff + j]; }

    // A fragments: W1^T / W2^T chunks, pre-swizzled
    const _Float16* pw1 = packedW + (size_t)((head * 2 + 0) * 2) * 32 * 16;
    const _Float16* pw2 = packedW + (size_t)((head * 2 + 1) * 2) * 32 * 16;
    v16h a1lo = *(const v16h*)(pw1 + (0 * 32 + lane) * 16);
    v16h a1hi = *(const v16h*)(pw1 + (1 * 32 + lane) * 16);
    v16h a2lo = *(const v16h*)(pw2 + (0 * 32 + lane) * 16);
    v16h a2hi = *(const v16h*)(pw2 + (1 * 32 + lane) * 16);

    // biases as WMMA C accumulators (C/D layout: row m = j + koff (+16 for hi))
    v8f c1lo, c1hi, c2lo, c2hi;
    #pragma unroll
    for (int j = 0; j < 8; ++j) {
        c1lo[j] = bias1[koff + j];
        c1hi[j] = bias1[16 + koff + j];
        c2lo[j] = bias2[koff + j];
        c2hi[j] = bias2[16 + koff + j];
    }
    float w3a[8], w3b[8];
    #pragma unroll
    for (int j = 0; j < 8; ++j) { w3a[j] = w3[koff + j]; w3b[j] = w3[16 + koff + j]; }

    const float* ipb = i_part + (size_t)head * N_ITEMS * HID;
    int item = blockIdx.y * ITEM_CHUNK + slot * 2;

    // preload first item pair
    float ca[16], cb[16];
    {
        const float* p0 = ipb + (size_t)item * HID;
        const float* p1 = p0 + HID;
        #pragma unroll
        for (int j = 0; j < 8; ++j) {
            ca[j] = p0[koff + j]; ca[8 + j] = p0[16 + koff + j];
            cb[j] = p1[koff + j]; cb[8 + j] = p1[16 + koff + j];
        }
    }

    #pragma unroll 1
    for (int t = 0; t < PAIR_ITERS; ++t) {
        // prefetch next pair (clamped on last iteration; stays in-bounds)
        const int nx = (t + 1 < PAIR_ITERS) ? (item + 8) : item;
        float na[16], nb[16];
        {
            const float* p0 = ipb + (size_t)nx * HID;
            const float* p1 = p0 + HID;
            #pragma unroll
            for (int j = 0; j < 8; ++j) {
                na[j] = p0[koff + j]; na[8 + j] = p0[16 + koff + j];
                nb[j] = p1[koff + j]; nb[8 + j] = p1[16 + koff + j];
            }
        }

        // H1^T in f16 B layout for both items (lane-local relu(u+i) pack)
        v16h bA, bB;
        #pragma unroll
        for (int j = 0; j < 16; ++j) {
            bA[j] = (_Float16)fmaxf(u[j] + ca[j], 0.f);
            bB[j] = (_Float16)fmaxf(u[j] + cb[j], 0.f);
        }
        // layer 2: 4 independent WMMAs
        v8f dA0 = __builtin_amdgcn_wmma_f32_16x16x32_f16(false, a1lo, false, bA, (short)0, c1lo, false, false);
        v8f dA1 = __builtin_amdgcn_wmma_f32_16x16x32_f16(false, a1hi, false, bA, (short)0, c1hi, false, false);
        v8f dB0 = __builtin_amdgcn_wmma_f32_16x16x32_f16(false, a1lo, false, bB, (short)0, c1lo, false, false);
        v8f dB1 = __builtin_amdgcn_wmma_f32_16x16x32_f16(false, a1hi, false, bB, (short)0, c1hi, false, false);
        // D layout -> next-layer B layout is lane-local
        v16h eA, eB;
        #pragma unroll
        for (int j = 0; j < 8; ++j) {
            eA[j]     = (_Float16)fmaxf(dA0[j], 0.f);
            eA[8 + j] = (_Float16)fmaxf(dA1[j], 0.f);
            eB[j]     = (_Float16)fmaxf(dB0[j], 0.f);
            eB[8 + j] = (_Float16)fmaxf(dB1[j], 0.f);
        }
        // layer 3: 4 independent WMMAs
        dA0 = __builtin_amdgcn_wmma_f32_16x16x32_f16(false, a2lo, false, eA, (short)0, c2lo, false, false);
        dA1 = __builtin_amdgcn_wmma_f32_16x16x32_f16(false, a2hi, false, eA, (short)0, c2hi, false, false);
        dB0 = __builtin_amdgcn_wmma_f32_16x16x32_f16(false, a2lo, false, eB, (short)0, c2lo, false, false);
        dB1 = __builtin_amdgcn_wmma_f32_16x16x32_f16(false, a2hi, false, eB, (short)0, c2hi, false, false);
        // layer 4: per-lane partial dots + xor-16 reduce
        float accA = 0.f, accB = 0.f;
        #pragma unroll
        for (int j = 0; j < 8; ++j) {
            accA += fmaxf(dA0[j], 0.f) * w3a[j] + fmaxf(dA1[j], 0.f) * w3b[j];
            accB += fmaxf(dB0[j], 0.f) * w3a[j] + fmaxf(dB1[j], 0.f) * w3b[j];
        }
        accA += __shfl_xor(accA, 16, 32);
        accB += __shfl_xor(accB, 16, 32);
        if (lane < 16) {
            float2 r;
            r.x = 1.f / (1.f + __expf(-(accA + b3)));
            r.y = 1.f / (1.f + __expf(-(accB + b3)));
            *(float2*)(out + (size_t)(utile + n) * N_ITEMS + item) = r;
        }
        // rotate double buffers
        #pragma unroll
        for (int j = 0; j < 16; ++j) { ca[j] = na[j]; cb[j] = nb[j]; }
        item += 8;
    }
}

// user_sim[i][j] = (1 - cos(eu_i, eu_j)) / 2
__global__ void k_usersim(const float* __restrict__ norm_eu, float* __restrict__ out) {
    int i = blockIdx.x, j = threadIdx.x;   // 256 x 256
    const float* a = norm_eu + i * D_EMB;
    const float* b = norm_eu + j * D_EMB;
    float dp = 0.f;
    #pragma unroll
    for (int d = 0; d < D_EMB; ++d) dp += a[d] * b[d];
    out[i * B_USERS + j] = 0.5f * (1.f - dp);
}

// popular head: plain per-thread MLP over 10000 items (negligible work)
__global__ void k_pop(const float* __restrict__ item_emb,
                      const float* __restrict__ w0, const float* __restrict__ b0,
                      const float* __restrict__ w1, const float* __restrict__ b1,
                      const float* __restrict__ w2, const float* __restrict__ b2,
                      const float* __restrict__ w3, const float* __restrict__ b3,
                      float* __restrict__ out) {
    int i = blockIdx.x * blockDim.x + threadIdx.x;
    if (i >= N_ITEMS) return;
    const float* x = item_emb + (size_t)i * D_EMB;
    float h1[HID], h2[HID];
    #pragma unroll 4
    for (int k = 0; k < HID; ++k) {
        float a = b0[k];
        for (int d = 0; d < D_EMB; ++d) a += x[d] * w0[d * HID + k];
        h1[k] = fmaxf(a, 0.f);
    }
    #pragma unroll 4
    for (int k = 0; k < HID; ++k) {
        float a = b1[k];
        for (int d = 0; d < HID; ++d) a += h1[d] * w1[d * HID + k];
        h2[k] = fmaxf(a, 0.f);
    }
    #pragma unroll 4
    for (int k = 0; k < HID; ++k) {
        float a = b2[k];
        for (int d = 0; d < HID; ++d) a += h2[d] * w2[d * HID + k];
        h1[k] = fmaxf(a, 0.f);   // reuse h1 as h3
    }
    float o = b3[0];
    #pragma unroll
    for (int k = 0; k < HID; ++k) o += h1[k] * w3[k];
    out[i] = 1.f / (1.f + __expf(-o));
}

extern "C" void kernel_launch(void* const* d_in, const int* in_sizes, int n_in,
                              void* d_out, int out_size, void* d_ws, size_t ws_size,
                              hipStream_t stream) {
    const int*   users    = (const int*)  d_in[0];
    const float* user_emb = (const float*)d_in[1];
    const float* item_emb = (const float*)d_in[2];
    const float* lw0 = (const float*)d_in[3],  *lb0 = (const float*)d_in[4];
    const float* lw1 = (const float*)d_in[5],  *lb1 = (const float*)d_in[6];
    const float* lw2 = (const float*)d_in[7],  *lb2 = (const float*)d_in[8];
    const float* lw3 = (const float*)d_in[9],  *lb3 = (const float*)d_in[10];
    const float* rw0 = (const float*)d_in[11], *rb0 = (const float*)d_in[12];
    const float* rw1 = (const float*)d_in[13], *rb1 = (const float*)d_in[14];
    const float* rw2 = (const float*)d_in[15], *rb2 = (const float*)d_in[16];
    const float* rw3 = (const float*)d_in[17], *rb3 = (const float*)d_in[18];
    const float* pw0 = (const float*)d_in[19], *pb0 = (const float*)d_in[20];
    const float* pw1 = (const float*)d_in[21], *pb1 = (const float*)d_in[22];
    const float* pw2 = (const float*)d_in[23], *pb2 = (const float*)d_in[24];
    const float* pw3 = (const float*)d_in[25], *pb3 = (const float*)d_in[26];

    char* ws = (char*)d_ws;
    float*    u_part  = (float*)(ws);
    float*    norm_eu = (float*)(ws + 65536);
    _Float16* packedW = (_Float16*)(ws + 131072);
    float*    i_part  = (float*)(ws + 139264);

    float* likes    = (float*)d_out;                      // [256][10000]
    float* user_sim = (float*)d_out + 2560000;            // [256][256]
    float* rated    = (float*)d_out + 2625536;            // [256][10000]
    float* popular  = (float*)d_out + 5185536;            // [10000]

    k_user_prep<<<64, 256, 0, stream>>>(users, user_emb, lw0, lb0, rw0, rb0, u_part);
    k_norm<<<1, 256, 0, stream>>>(users, user_emb, norm_eu);
    k_ipart<<<2500, 256, 0, stream>>>(item_emb, lw0, rw0, i_part);
    k_pack<<<1, 256, 0, stream>>>(lw1, lw2, rw1, rw2, packedW);

    k_pair<<<dim3(16, N_ITEMS / ITEM_CHUNK), 256, 0, stream>>>(
        u_part, i_part, packedW,
        lb1, lb2, lw3, lb3, rb1, rb2, rw3, rb3,
        likes, rated);

    k_usersim<<<B_USERS, B_USERS, 0, stream>>>(norm_eu, user_sim);
    k_pop<<<(N_ITEMS + 255) / 256, 256, 0, stream>>>(item_emb, pw0, pb0, pw1, pb1,
                                                     pw2, pb2, pw3, pb3, popular);
}
